// GRU_74603581931739
// MI455X (gfx1250) — compile-verified, matrix operand order checked by reading, not run on hardware
//
#include <hip/hip_runtime.h>
#include <hip/hip_bf16.h>

// GRU: B=64, T=512, I=256, H=1024.  (MI455X / gfx1250, wave32, WMMA bf16)
//  - prep: convert w_ih/w_hh -> bf16 in ws, transpose+convert x -> xT[T][B][I] bf16,
//    zero h0 and the grid-barrier counter.
//  - persistent kernel runs all 512 timesteps. 64 blocks x 384 threads (12 wave32).
//    Block b owns H-columns [16b,16b+16). Wave (gate g, mtile) computes a 16x16 f32 tile
//    of gi+gh via v_wmma_f32_16x16x32_bf16 (8 gi chunks K=256 + 32 gh chunks K=1024).
//  - h staged into LDS via GLOBAL_LOAD_ASYNC_TO_LDS_B128 (ASYNCcnt), overlapped with the
//    gi GEMM which is independent of h; then s_wait_asynccnt + barrier; then gh GEMM.
//  - K loops use `#pragma unroll 4` (keeps ~4 frag pairs in flight, <256 VGPRs, no spills).
//  - Grid-wide step barrier: monotonic atomic counter + s_sleep spin (64 resident blocks).

#define B_   64
#define T_   512
#define I_   256
#define H_   1024
#define G3_  3072
#define NBLK 64
#define NTHR 384
#define HPAD 1032        // h_lds row stride: 2064B -> 16B aligned, 4-bank shift/row

typedef __bf16 bf16x16 __attribute__((ext_vector_type(16)));
typedef float  f32x8   __attribute__((ext_vector_type(8)));

union FragA { bf16x16 v; uint4 q[2]; };

__device__ __forceinline__ bf16x16 load_frag(const unsigned short* row, int kb) {
    FragA f;
    f.q[0] = *reinterpret_cast<const uint4*>(row + kb);        // K = kb .. kb+7
    f.q[1] = *reinterpret_cast<const uint4*>(row + kb + 16);   // K = kb+16 .. kb+23
    return f.v;
}

__device__ __forceinline__ unsigned short f2bf(float f) {
    unsigned u = __float_as_uint(f);
    unsigned r = (u + 0x7FFFu + ((u >> 16) & 1u)) >> 16;       // RNE
    return (unsigned short)r;
}
__device__ __forceinline__ float bf2f(unsigned short h) {
    return __uint_as_float(((unsigned)h) << 16);
}
__device__ __forceinline__ float sigmoidf_(float x) {
    return 1.0f / (1.0f + __expf(-x));
}

// ---------------- prep: weight conversion + h0/ctr init ----------------
__global__ void gru_prep_kernel(const float* __restrict__ w_ih,
                                const float* __restrict__ w_hh,
                                unsigned short* __restrict__ wihb,
                                unsigned short* __restrict__ whhb,
                                unsigned short* __restrict__ h0,
                                unsigned int*   __restrict__ ctr) {
    const int stride = gridDim.x * blockDim.x;
    int tid = blockIdx.x * blockDim.x + threadIdx.x;
    if (tid == 0) *ctr = 0u;
    for (int i = tid; i < G3_ * H_; i += stride) whhb[i] = f2bf(w_hh[i]);
    for (int i = tid; i < G3_ * I_; i += stride) wihb[i] = f2bf(w_ih[i]);
    for (int i = tid; i < B_ * H_;  i += stride) h0[i] = 0;
}

// ---------------- prep: x [B][I][T] f32 -> xT [T][B][I] bf16 (tiled transpose) --------
__global__ void gru_xpose_kernel(const float* __restrict__ x,
                                 unsigned short* __restrict__ xT) {
    __shared__ float tile[32][33];
    const int t0 = blockIdx.x * 32;
    const int i0 = blockIdx.y * 32;
    const int b  = blockIdx.z;
    const int tx = threadIdx.x;          // 0..31
    const int ty = threadIdx.y;          // 0..7
#pragma unroll
    for (int r = 0; r < 4; ++r) {
        int i = i0 + ty + r * 8;
        tile[ty + r * 8][tx] = x[((size_t)b * I_ + i) * T_ + t0 + tx];   // coalesced over t
    }
    __syncthreads();
#pragma unroll
    for (int r = 0; r < 4; ++r) {
        int t = t0 + ty + r * 8;
        xT[((size_t)t * B_ + b) * I_ + i0 + tx] = f2bf(tile[tx][ty + r * 8]); // coalesced over i
    }
}

// ---------------- persistent recurrent kernel ----------------
__global__ void __launch_bounds__(NTHR, 1)
gru_persistent_kernel(const unsigned short* __restrict__ xT,    // [T][B][I] bf16
                      const unsigned short* __restrict__ wihb,  // [3H][I]   bf16
                      const unsigned short* __restrict__ whhb,  // [3H][H]   bf16
                      const float* __restrict__ b_ih,           // [3H]
                      const float* __restrict__ b_hh,           // [3H]
                      unsigned short* __restrict__ h_buf0,      // [B][H] bf16 ping
                      unsigned short* __restrict__ h_buf1,      // [B][H] bf16 pong
                      unsigned int*   __restrict__ ctr,
                      float* __restrict__ out,                  // [B][T][H]
                      float* __restrict__ hid) {                // [B][H]
    __shared__ __align__(16) unsigned short h_lds[B_][HPAD];    // 132 KB
    // stage: 0=r(gi+gh), 1=z(gi+gh), 2=i_n, 3=h_n ; [gate][mtile][M][N] f32 (16 KB)
    __shared__ __align__(16) float stage[4][4][16][16];

    const int tid  = threadIdx.x;
    const int lane = tid & 31;
    const int wid  = tid >> 5;         // 0..11
    const int g    = wid >> 2;         // gate: 0=r 1=z 2=n
    const int mt   = wid & 3;          // M tile (batch rows mt*16..+15)
    const int lrow = lane & 15;        // A: M row / B: N col within tile
    const int khi  = lane >> 4;        // K-group select
    const int hc0  = blockIdx.x * 16;  // this block's H column base

    const int jrow = g * H_ + hc0 + lrow;                    // gate row in [3H]
    const unsigned short* whh_row = whhb + (size_t)jrow * H_;
    const unsigned short* wih_row = wihb + (size_t)jrow * I_;

    // LDS byte offset of h_lds base (generic LDS pointer: low 32 bits = LDS address)
    const unsigned h_lds_base = (unsigned)(uintptr_t)&h_lds[0][0];

    for (int t = 0; t < T_; ++t) {
        const unsigned short* h_cur = (t & 1) ? h_buf1 : h_buf0;
        unsigned short*       h_nxt = (t & 1) ? h_buf0 : h_buf1;

        // prefetch next timestep's x slice into L2
        if (t + 1 < T_)
            __builtin_prefetch(xT + ((size_t)(t + 1) * B_ + (tid & 63)) * I_, 0, 1);

        __syncthreads();   // prior users of h_lds / stage done

        // ---- kick off async DMA of full h [64][1024] bf16 into LDS (8192 x b128) ----
        {
            const unsigned long long gbase = (unsigned long long)(uintptr_t)h_cur;
            for (int idx = tid; idx < B_ * 128; idx += NTHR) {
                int row = idx >> 7, c = idx & 127;
                unsigned long long ga = gbase + (unsigned)(row * 2048 + c * 16);
                unsigned ld = h_lds_base + (unsigned)(row * (HPAD * 2) + c * 16);
                asm volatile("global_load_async_to_lds_b128 %0, %1, off"
                             :: "v"(ld), "v"(ga) : "memory");
            }
        }

        f32x8 acc_hh = {};   // gh accumulation (K=1024)
        f32x8 acc_ih = {};   // gi accumulation (K=256)

        // ---- gi = x_t @ w_ih^T : 8 chunks of K=32, overlapped with the async DMA ----
        {
            const unsigned short* a_row = xT + ((size_t)t * B_ + mt * 16 + lrow) * I_;
#pragma unroll 4
            for (int kc = 0; kc < 8; ++kc) {
                int kb = kc * 32 + khi * 8;
                bf16x16 a = load_frag(a_row, kb);
                bf16x16 b = load_frag(wih_row, kb);
                acc_ih = __builtin_amdgcn_wmma_f32_16x16x32_bf16(
                    false, a, false, b, (short)0, acc_ih, false, false);
            }
        }

        // ---- wait for my async transfers, then barrier so all waves' data is visible ----
        asm volatile("s_wait_asynccnt 0x0" ::: "memory");
        __syncthreads();

        // ---- gh = h @ w_hh^T : 32 chunks of K=32 out of LDS ----
        {
            const unsigned short* a_row = &h_lds[mt * 16 + lrow][0];
#pragma unroll 4
            for (int kc = 0; kc < 32; ++kc) {
                int kb = kc * 32 + khi * 8;
                bf16x16 a = load_frag(a_row, kb);
                bf16x16 b = load_frag(whh_row, kb);
                acc_hh = __builtin_amdgcn_wmma_f32_16x16x32_bf16(
                    false, a, false, b, (short)0, acc_hh, false, false);
            }
        }

        // ---- stage tiles: C/D layout: VGPR v -> M = v + 8*khi, lane -> N ----
#pragma unroll
        for (int v = 0; v < 8; ++v) {
            int row = khi * 8 + v;
            if (g < 2) {
                stage[g][mt][row][lrow] = acc_hh[v] + acc_ih[v];
            } else {
                stage[2][mt][row][lrow] = acc_ih[v];   // i_n
                stage[3][mt][row][lrow] = acc_hh[v];   // h_n
            }
        }
        __syncthreads();

        // ---- gates + h update for this block's 64x16 slab ----
        for (int e = tid; e < B_ * 16; e += NTHR) {
            int m = e >> 4, c = e & 15;
            int mti = m >> 4, mr = m & 15;
            int hcol = hc0 + c;
            float vr = stage[0][mti][mr][c] + b_ih[hcol] + b_hh[hcol];
            float vz = stage[1][mti][mr][c] + b_ih[H_ + hcol] + b_hh[H_ + hcol];
            float vi = stage[2][mti][mr][c] + b_ih[2 * H_ + hcol];
            float vh = stage[3][mti][mr][c] + b_hh[2 * H_ + hcol];
            float r = sigmoidf_(vr);
            float z = sigmoidf_(vz);
            float n = tanhf(vi + r * vh);
            float hold = bf2f(h_cur[(size_t)m * H_ + hcol]);
            float hnew = (1.0f - z) * n + z * hold;
            h_nxt[(size_t)m * H_ + hcol] = f2bf(hnew);
            out[((size_t)m * T_ + t) * H_ + hcol] = hnew;
            if (t == T_ - 1) hid[(size_t)m * H_ + hcol] = hnew;
        }

        // ---- grid-wide step barrier (monotonic counter; 64 resident blocks) ----
        if (t != T_ - 1) {
            __threadfence();
            __syncthreads();
            if (tid == 0) {
                __hip_atomic_fetch_add(ctr, 1u, __ATOMIC_RELEASE, __HIP_MEMORY_SCOPE_AGENT);
                unsigned target = (unsigned)(t + 1) * gridDim.x;
                while (__hip_atomic_load(ctr, __ATOMIC_ACQUIRE, __HIP_MEMORY_SCOPE_AGENT) < target)
                    __builtin_amdgcn_s_sleep(2);
            }
            __syncthreads();
        }
    }
}

// ---------------- launch ----------------
extern "C" void kernel_launch(void* const* d_in, const int* in_sizes, int n_in,
                              void* d_out, int out_size, void* d_ws, size_t ws_size,
                              hipStream_t stream) {
    (void)in_sizes; (void)n_in; (void)out_size; (void)ws_size;
    const float* x    = (const float*)d_in[0];   // [64][256][512]
    const float* w_ih = (const float*)d_in[1];   // [3072][256]
    const float* w_hh = (const float*)d_in[2];   // [3072][1024]
    const float* b_ih = (const float*)d_in[3];   // [3072]
    const float* b_hh = (const float*)d_in[4];   // [3072]

    char* ws = (char*)d_ws;
    unsigned int*   ctr   = (unsigned int*)  (ws + 0);
    unsigned short* h0    = (unsigned short*)(ws + 256);                       // 128 KB
    unsigned short* h1    = (unsigned short*)(ws + 256 + 131072);              // 128 KB
    unsigned short* wihb  = (unsigned short*)(ws + 262400);                    // 1.5 MB
    unsigned short* whhb  = (unsigned short*)(ws + 1835264);                   // 6 MB
    unsigned short* xT    = (unsigned short*)(ws + 8126720);                   // 16 MB

    float* out = (float*)d_out;
    float* hid = out + (size_t)B_ * T_ * H_;

    gru_prep_kernel<<<2048, 256, 0, stream>>>(w_ih, w_hh, wihb, whhb, h0, ctr);
    gru_xpose_kernel<<<dim3(T_ / 32, I_ / 32, B_), dim3(32, 8), 0, stream>>>(x, xT);
    gru_persistent_kernel<<<NBLK, NTHR, 0, stream>>>(xT, wihb, whhb, b_ih, b_hh,
                                                     h0, h1, ctr, out, hid);
}
